// CriticNetwork_11811160064541
// MI455X (gfx1250) — matrix operand order repaired and checked by reading, not compiled, not run on hardware
//
#include <hip/hip_runtime.h>
#include <hip/hip_bf16.h>
#include <math.h>

typedef __attribute__((ext_vector_type(16))) _Float16 v16h;
typedef __attribute__((ext_vector_type(8)))  _Float16 v8h;
typedef __attribute__((ext_vector_type(8)))  float    v8f;

#define LRELU 0.01f

// ---------------- LDS layout (offsets in halves) ----------------
// obsz (256x96) is later aliased by node (256x128); hid head also aliased by node.
#define OBSZ_OFF 0        // 24576 halves
#define NODE_OFF 0        // 32768 halves (aliases OBSZ + head of HID, both dead)
#define HID_OFF  24576    // 256*64 = 16384
#define BUF_OFF  40960    // 256*64 = 16384
#define OBS_OFF  57344    // 16*64
#define H1_OFF   58368    // 16*64
#define H2_OFF   59392    // 16*64
#define KEYZ_OFF 60416    // 16*64
#define QZ_OFF   61440    // 16*64
#define KEYO_OFF 62464    // 16*64
#define AOP_OFF  63488    // 16*64
#define OP_OFF   64512    // 16*96
#define WO16_OFF 66048    // 16*32
#define F32_OFF  66560    // float region: w(256) y(256) wo(256) pi(128) act(128)
#define SMEM_HALVES 68608 // ~134 KB static LDS (gfx1250 WGP: 320 KB)

#define WS_HALVES 56320   // total f16 weight workspace

struct KParams {
  const float *obs, *pol, *act;
  const _Float16 *kw1,*kw2,*qw1,*qw2,*ko1,*ko2,*qo1,*qo2,*av1,*av2,*fv1,*fv2;
  const float *kw1b,*kw2b,*qw1b,*qw2b,*ko1b,*ko2b,*qo1b,*qo2b,*av1b,*av2b,*fv1b,*fv2b;
  float *o_val, *o_wz, *o_wobsz;
};

__device__ __forceinline__ v16h cat8(v8h lo, v8h hi) {
  return __builtin_shufflevector(lo, hi, 0,1,2,3,4,5,6,7,8,9,10,11,12,13,14,15);
}

// A-fragment: 16xK tile, row m=lane%16; K halves: g*16+[0..15]
__device__ __forceinline__ v16h load_a_frag(const _Float16* base, int row_base,
                                            int row_stride, int k_base, int lane) {
  int g = (lane >> 4) & 1;
  int m = lane & 15;
  const _Float16* p = base + (size_t)(row_base + m) * row_stride + k_base + g * 16;
  v8h lo = *(const v8h*)(p);
  v8h hi = *(const v8h*)(p + 8);
  return cat8(lo, hi);
}

// B-fragment for x@W^T: B[k][n] = W[n][k]; lane holds col n=lane%16,
// K = g*16 + [0..15]  -> 16 contiguous halves of row n of W.
__device__ __forceinline__ v16h load_b_frag(const _Float16* W, int n_base,
                                            int k_stride, int k_base, int lane) {
  int g = (lane >> 4) & 1;
  int n = lane & 15;
  const _Float16* p = W + (size_t)(n_base + n) * k_stride + k_base + g * 16;
  v8h lo = *(const v8h*)(p);
  v8h hi = *(const v8h*)(p + 8);
  return cat8(lo, hi);
}

// One MLP-layer GEMM: out = act(in @ W^T + b), M = m_tiles*16 rows, N = n_tiles*16,
// K = kc*32 (zero padded).  Tiles distributed over waves.
__device__ __forceinline__ void gemm_layer(
    const _Float16* in, int in_stride, int m_tiles,
    const _Float16* W, int w_stride,
    const float* bias, int n_tiles, int kc,
    _Float16* out, int out_stride, bool act,
    int wave, int nwaves, int lane)
{
  int g = lane >> 4, n = lane & 15;
  for (int t = wave; t < m_tiles * n_tiles; t += nwaves) {
    int mt = t / n_tiles, nt = t - mt * n_tiles;
    v8f c = {};
    for (int k = 0; k < kc; ++k) {
      v16h a = load_a_frag(in, mt * 16, in_stride, k * 32, lane);
      v16h b = load_b_frag(W, nt * 16, w_stride, k * 32, lane);
      c = __builtin_amdgcn_wmma_f32_16x16x32_f16(false, a, false, b, (short)0, c,
                                                 false, false);
    }
    float bv = bias[nt * 16 + n];
#pragma unroll
    for (int r = 0; r < 8; ++r) {
      int m = r + 8 * g;
      float v = c[r] + bv;
      if (act) v = v > 0.f ? v : LRELU * v;
      out[(size_t)(mt * 16 + m) * out_stride + nt * 16 + n] = (_Float16)v;
    }
  }
}

// ---- single-launch weight conversion: f32 -> f16, zero-padded rows/cols ----
struct CvtDesc { const float* src; _Float16* dst; int srows, scols, drows, kpad; };
struct CvtParams { CvtDesc d[12]; };

__global__ __launch_bounds__(256)
void cvt_all(CvtParams C) {
  CvtDesc d = C.d[blockIdx.x];
  int total = d.drows * d.kpad;
  for (int idx = threadIdx.x; idx < total; idx += 256) {
    int r = idx / d.kpad, c = idx - r * d.kpad;
    float v = (r < d.srows && c < d.scols) ? d.src[r * d.scols + c] : 0.f;
    d.dst[idx] = (_Float16)v;
  }
}

__global__ __launch_bounds__(256)
void critic_kernel(KParams P) {
  __shared__ __align__(16) _Float16 smem[SMEM_HALVES];
  _Float16* s_obsz = smem + OBSZ_OFF;
  _Float16* s_node = smem + NODE_OFF;
  _Float16* s_hid  = smem + HID_OFF;
  _Float16* s_buf  = smem + BUF_OFF;
  _Float16* s_obs  = smem + OBS_OFF;
  _Float16* s_h1   = smem + H1_OFF;
  _Float16* s_h2   = smem + H2_OFF;
  _Float16* s_keyz = smem + KEYZ_OFF;
  _Float16* s_qz   = smem + QZ_OFF;
  _Float16* s_keyo = smem + KEYO_OFF;
  _Float16* s_aop  = smem + AOP_OFF;
  _Float16* s_op   = smem + OP_OFF;
  _Float16* s_wo16 = smem + WO16_OFF;
  float* fr   = (float*)(smem + F32_OFF);
  float* s_w  = fr;          // sigmoid weights  16x16
  float* s_y  = fr + 256;    // exp(clip(score_o/8))
  float* s_wo = fr + 512;    // softmax weights  16x16
  float* s_pi = fr + 768;    // 16x8
  float* s_ac = fr + 896;    // 16x8

  const int tid  = threadIdx.x;
  const int lane = tid & 31;
  const int wave = tid >> 5;
  const int b    = blockIdx.x;

  // ---- warm L2 for the shared f16 weights (global_prefetch_b8) ----
  {
    const char* wp = (const char*)P.kw1;   // start of weight workspace
    const int nlines = (WS_HALVES * 2) / 256;  // 256B granules
    for (int q = tid; q < nlines; q += 256)
      __builtin_prefetch(wp + q * 256, 0, 1);
  }

  // ---- stage 0: load obs / pi / act ----
  {
    const float* ob = P.obs + (size_t)b * 1024;
#pragma unroll
    for (int r = 0; r < 4; ++r) {
      int idx = tid + r * 256;
      s_obs[idx] = (_Float16)ob[idx];
    }
    if (tid < 128) {
      s_pi[tid] = P.pol[(size_t)b * 128 + tid];
      s_ac[tid] = P.act[(size_t)b * 128 + tid];
    }
  }
  __syncthreads();

  // ---- stage 1: key_z / query_z MLPs (16 rows each, run on disjoint wave halves)
  if (wave < 4) gemm_layer(s_obs, 64, 1, P.kw1, 64, P.kw1b, 4, 2, s_h1, 64, true,  wave,   4, lane);
  else          gemm_layer(s_obs, 64, 1, P.qw1, 64, P.qw1b, 4, 2, s_h2, 64, true,  wave-4, 4, lane);
  __syncthreads();
  if (wave < 4) gemm_layer(s_h1, 64, 1, P.kw2, 64, P.kw2b, 4, 2, s_keyz, 64, false, wave,   4, lane);
  else          gemm_layer(s_h2, 64, 1, P.qw2, 64, P.qw2b, 4, 2, s_qz,   64, false, wave-4, 4, lane);
  __syncthreads();

  // ---- stage 2: score = query_z @ key_z^T ; w = sigmoid(score/8); emit weight_z
  if (wave == 0) {
    v8f c = {};
    for (int k = 0; k < 2; ++k) {
      v16h a  = load_a_frag(s_qz,   0, 64, k * 32, lane);
      v16h bb = load_b_frag(s_keyz, 0, 64, k * 32, lane);
      c = __builtin_amdgcn_wmma_f32_16x16x32_f16(false, a, false, bb, (short)0, c,
                                                 false, false);
    }
    int g = lane >> 4, n = lane & 15;
#pragma unroll
    for (int r = 0; r < 8; ++r) {
      int i = r + 8 * g;
      float w = 1.f / (1.f + __expf(-c[r] * 0.125f));
      s_w[i * 16 + n] = w;
      P.o_wz[(size_t)b * 256 + i * 16 + n] = w;
    }
  }
  __syncthreads();

  // ---- stage 3: build obs_z rows (256 x 96, padded) and op rows (16 x 96)
  {
    int j = tid & 15;
    _Float16* row = s_obsz + (size_t)tid * 96;
    const uint4* sv = (const uint4*)(s_obs + j * 64);
    uint4* dv = (uint4*)row;
#pragma unroll
    for (int q = 0; q < 8; ++q) dv[q] = sv[q];
    float w = s_w[tid];
#pragma unroll
    for (int a = 0; a < 8; ++a)
      row[64 + a] = (_Float16)(w * s_ac[j * 8 + a] + (1.f - w) * s_pi[j * 8 + a]);
#pragma unroll
    for (int a = 72; a < 96; ++a) row[a] = (_Float16)0.f;
    if (tid < 16) {
      _Float16* orow = s_op + (size_t)tid * 96;
      const uint4* sv2 = (const uint4*)(s_obs + tid * 64);
      uint4* dv2 = (uint4*)orow;
#pragma unroll
      for (int q = 0; q < 8; ++q) dv2[q] = sv2[q];
#pragma unroll
      for (int a = 0; a < 8; ++a) orow[64 + a] = (_Float16)s_pi[tid * 8 + a];
#pragma unroll
      for (int a = 72; a < 96; ++a) orow[a] = (_Float16)0.f;
    }
  }
  __syncthreads();

  // ---- stage 4: query_o = MLP(obs_z)  (256 rows, K=96 padded)
  gemm_layer(s_obsz, 96, 16, P.qo1, 96, P.qo1b, 4, 3, s_hid, 64, true,  wave, 8, lane);
  __syncthreads();
  gemm_layer(s_hid,  64, 16, P.qo2, 64, P.qo2b, 4, 2, s_buf, 64, false, wave, 8, lane);
  __syncthreads();

  // ---- stage 5: key_o = MLP(src) (diag rows via stride 17*96), aop = MLP(op)
  if (wave < 4) gemm_layer(s_obsz, 17 * 96, 1, P.ko1, 96, P.ko1b, 4, 3, s_h1, 64, true,  wave,   4, lane);
  else          gemm_layer(s_op,   96,      1, P.av1, 96, P.av1b, 4, 3, s_h2, 64, true,  wave-4, 4, lane);
  __syncthreads();
  if (wave < 4) gemm_layer(s_h1, 64, 1, P.ko2, 64, P.ko2b, 4, 2, s_keyo, 64, false, wave,   4, lane);
  else          gemm_layer(s_h2, 64, 1, P.av2, 64, P.av2b, 4, 2, s_aop,  64, false, wave-4, 4, lane);
  __syncthreads();

  // ---- stage 6: score_o -> y = exp(clip(s/8)); softmax(y); emit weights_obsz
  {
    int i = tid >> 4;
    float s = 0.f;
    for (int d = 0; d < 64; ++d)
      s += (float)s_keyo[i * 64 + d] * (float)s_buf[(size_t)tid * 64 + d];
    s *= 0.125f;
    s = fminf(fmaxf(s, -5.f), 5.f);
    s_y[tid] = __expf(s);
  }
  __syncthreads();
  {
    int i = tid >> 4, j = tid & 15;
    float m = -1e30f;
    for (int jj = 0; jj < 16; ++jj) m = fmaxf(m, s_y[i * 16 + jj]);
    float sum = 0.f;
    for (int jj = 0; jj < 16; ++jj) sum += __expf(s_y[i * 16 + jj] - m);
    float wo = __expf(s_y[tid] - m) / sum;
    s_wo[tid] = wo;
    s_wo16[i * 32 + j]      = (_Float16)wo;
    s_wo16[i * 32 + 16 + j] = (_Float16)0.f;
  }
  __syncthreads();
  {
    int i = tid >> 4, j = tid & 15;
    float* o3 = P.o_wobsz + (size_t)b * 4096 + i * 256 + j;
#pragma unroll
    for (int k = 0; k < 16; ++k) o3[k * 16] = s_wo[k * 16 + j];
  }
  __syncthreads();

  // ---- stage 7: attn = MLP_av(obs_z)  (collapsed attn_other; diag == attn_src)
  gemm_layer(s_obsz, 96, 16, P.av1, 96, P.av1b, 4, 3, s_hid, 64, true,  wave, 8, lane);
  __syncthreads();
  gemm_layer(s_hid,  64, 16, P.av2, 64, P.av2b, 4, 2, s_buf, 64, false, wave, 8, lane);
  __syncthreads();

  // ---- stage 8: node = [attn_src | mixed]; mixed via WMMA  W_o(16x16) @ attn_i
  {
    int i0 = tid >> 4;
    const uint4* sv = (const uint4*)(s_buf + (size_t)i0 * 17 * 64);  // attn diag row
    uint4* dv = (uint4*)(s_node + (size_t)tid * 128);
#pragma unroll
    for (int q = 0; q < 8; ++q) dv[q] = sv[q];
  }
  for (int t = wave; t < 64; t += 8) {
    int i = t >> 2, nt = t & 3;
    v16h a = load_a_frag(s_wo16, 0, 32, 0, lane);   // W_o padded to 16x32
    int g = lane >> 4, n = lane & 15;
    v16h bb;
#pragma unroll
    for (int q = 0; q < 16; ++q) {
      _Float16 val = (_Float16)0.f;                 // K=16..31 is zero pad
      if (!g) val = s_buf[(size_t)(i * 16 + q) * 64 + nt * 16 + n];
      bb[q] = val;
    }
    v8f c = {};
    c = __builtin_amdgcn_wmma_f32_16x16x32_f16(false, a, false, bb, (short)0, c,
                                               false, false);
    int d = nt * 16 + n;
#pragma unroll
    for (int r = 0; r < 8; ++r) {
      int k = r + 8 * g;
      float wkk  = s_wo[k * 16 + k];
      float corr = wkk * ((float)s_aop[k * 64 + d] -
                          (float)s_buf[(size_t)(i * 16 + k) * 64 + d]);
      float v = (c[r] + corr) * 0.0625f;            // mean over 16
      s_node[(size_t)(i * 16 + k) * 128 + 64 + d] = (_Float16)v;
    }
  }
  __syncthreads();

  // ---- stage 9: value head: 256x128 -> 64 (lrelu) -> 8 (direct global store)
  gemm_layer(s_node, 128, 16, P.fv1, 128, P.fv1b, 4, 4, s_buf, 64, true, wave, 8, lane);
  __syncthreads();
  for (int t = wave; t < 16; t += 8) {
    v8f c = {};
    for (int k = 0; k < 2; ++k) {
      v16h a  = load_a_frag(s_buf, t * 16, 64, k * 32, lane);
      v16h bb = load_b_frag(P.fv2, 0, 64, k * 32, lane);  // padded to 16 rows
      c = __builtin_amdgcn_wmma_f32_16x16x32_f16(false, a, false, bb, (short)0, c,
                                                 false, false);
    }
    int g = lane >> 4, n = lane & 15;
    if (n < 8) {
      float bv = P.fv2b[n];
#pragma unroll
      for (int r = 0; r < 8; ++r) {
        int m = r + 8 * g;
        P.o_val[((size_t)b * 256 + t * 16 + m) * 8 + n] = c[r] + bv;
      }
    }
  }
}

extern "C" void kernel_launch(void* const* d_in, const int* in_sizes, int n_in,
                              void* d_out, int out_size, void* d_ws, size_t ws_size,
                              hipStream_t stream) {
  const int Bn = in_sizes[0] / 1024;  // (B*A*D_OBS) / (A*D_OBS) = 128
  _Float16* ws = (_Float16*)d_ws;

  // f16 weight workspace offsets (halves)
  const size_t KW1 = 0, KW2 = 4096, QW1 = 8192, QW2 = 12288;
  const size_t KO1 = 16384, KO2 = 22528, QO1 = 26624, QO2 = 32768;
  const size_t AV1 = 36864, AV2 = 43008, FV1 = 47104, FV2 = 55296;

  CvtParams C;
  auto set = [&](int slot, int i, size_t off, int sr, int sc, int dr, int kp) {
    C.d[slot] = CvtDesc{(const float*)d_in[i], ws + off, sr, sc, dr, kp};
  };
  set(0,  3,  KW1, 64, 64,  64, 64);
  set(1,  5,  KW2, 64, 64,  64, 64);
  set(2,  7,  QW1, 64, 64,  64, 64);
  set(3,  9,  QW2, 64, 64,  64, 64);
  set(4,  11, KO1, 64, 72,  64, 96);
  set(5,  13, KO2, 64, 64,  64, 64);
  set(6,  15, QO1, 64, 72,  64, 96);
  set(7,  17, QO2, 64, 64,  64, 64);
  set(8,  19, AV1, 64, 72,  64, 96);
  set(9,  21, AV2, 64, 64,  64, 64);
  set(10, 23, FV1, 64, 128, 64, 128);
  set(11, 25, FV2, 8,  64,  16, 64);
  hipLaunchKernelGGL(cvt_all, dim3(12), dim3(256), 0, stream, C);

  KParams P;
  P.obs = (const float*)d_in[0];
  P.pol = (const float*)d_in[1];
  P.act = (const float*)d_in[2];
  P.kw1 = ws + KW1;  P.kw2 = ws + KW2;  P.qw1 = ws + QW1;  P.qw2 = ws + QW2;
  P.ko1 = ws + KO1;  P.ko2 = ws + KO2;  P.qo1 = ws + QO1;  P.qo2 = ws + QO2;
  P.av1 = ws + AV1;  P.av2 = ws + AV2;  P.fv1 = ws + FV1;  P.fv2 = ws + FV2;
  P.kw1b = (const float*)d_in[4];   P.kw2b = (const float*)d_in[6];
  P.qw1b = (const float*)d_in[8];   P.qw2b = (const float*)d_in[10];
  P.ko1b = (const float*)d_in[12];  P.ko2b = (const float*)d_in[14];
  P.qo1b = (const float*)d_in[16];  P.qo2b = (const float*)d_in[18];
  P.av1b = (const float*)d_in[20];  P.av2b = (const float*)d_in[22];
  P.fv1b = (const float*)d_in[24];  P.fv2b = (const float*)d_in[26];

  float* out = (float*)d_out;
  P.o_val   = out;
  P.o_wz    = out + (size_t)Bn * 2048;
  P.o_wobsz = out + (size_t)Bn * 2048 + (size_t)Bn * 256;

  hipLaunchKernelGGL(critic_kernel, dim3(Bn), dim3(256), 0, stream, P);
}
